// E2ETTT_20736102105397
// MI455X (gfx1250) — matrix operand order repaired and checked by reading, not compiled, not run on hardware
//
#include <hip/hip_runtime.h>
#include <math.h>
#include <stdint.h>

// ---------------- model constants ----------------
#define TT   1024      // sequence length T
#define BB   2         // batch
#define DD   1024      // model dim
#define NH   16        // heads
#define HDIM 64        // head dim
#define NL   8         // layers
#define HALFF 2048     // FF/2
#define WIN  256       // attention window
#define VV   32000     // vocab
#define MTOK (BB*TT)   // 2048 tokens

typedef __attribute__((ext_vector_type(16))) __bf16 bf16x16;
typedef __attribute__((ext_vector_type(8)))  float  f32x8;
typedef __attribute__((ext_vector_type(4)))  unsigned int su32x4;
typedef __attribute__((ext_vector_type(8)))  unsigned int su32x8;

union BF16Frag { uint4 u[2]; bf16x16 v; };

__device__ __forceinline__ unsigned short f2bf(float f) {
    unsigned int u = __float_as_uint(f);
    u += 0x7FFFu + ((u >> 16) & 1u);      // round-to-nearest-even
    return (unsigned short)(u >> 16);
}

// ---------------- embedding: x = tok_emb[idx] + pos_emb ----------------
__global__ __launch_bounds__(256)
void embed_kernel(const int* __restrict__ idx, const float* __restrict__ tok,
                  const float* __restrict__ pos, float* __restrict__ x)
{
    int t = blockIdx.x;                // 0..2047 flat (b,t)
    int tid = threadIdx.x;             // 256 threads * float4 = 1024
    int token = idx[t];
    float4 te = ((const float4*)(tok + (size_t)token * DD))[tid];
    float4 pe = ((const float4*)(pos + (size_t)(t & (TT - 1)) * DD))[tid];
    float4 o;
    o.x = te.x + pe.x; o.y = te.y + pe.y; o.z = te.z + pe.z; o.w = te.w + pe.w;
    ((float4*)(x + (size_t)t * DD))[tid] = o;
}

// ---------------- bulk f32 -> bf16 weight conversion ----------------
__global__ __launch_bounds__(256)
void cvt_kernel(const float* __restrict__ src, unsigned short* __restrict__ dst)
{
    size_t i = ((size_t)blockIdx.x * 256 + threadIdx.x) * 8;
    float4 a = *(const float4*)(src + i);
    float4 b = *(const float4*)(src + i + 4);
    ushort4 o0, o1;
    o0.x = f2bf(a.x); o0.y = f2bf(a.y); o0.z = f2bf(a.z); o0.w = f2bf(a.w);
    o1.x = f2bf(b.x); o1.y = f2bf(b.y); o1.z = f2bf(b.z); o1.w = f2bf(b.w);
    *(ushort4*)(dst + i)     = o0;
    *(ushort4*)(dst + i + 4) = o1;
}

// ---------------- layernorm -> bf16 ----------------
__global__ __launch_bounds__(256)
void ln_kernel(const float* __restrict__ x, const float* __restrict__ g,
               const float* __restrict__ bta, unsigned short* __restrict__ out)
{
    int t = blockIdx.x;
    int tid = threadIdx.x;
    int lane = tid & 31, w = tid >> 5;
    float4 v = ((const float4*)(x + (size_t)t * DD))[tid];
    float s  = v.x + v.y + v.z + v.w;
    float s2 = v.x*v.x + v.y*v.y + v.z*v.z + v.w*v.w;
    #pragma unroll
    for (int off = 16; off >= 1; off >>= 1) {
        s  += __shfl_xor(s,  off, 32);
        s2 += __shfl_xor(s2, off, 32);
    }
    __shared__ float red[16];
    __shared__ float stat[2];
    if (lane == 0) { red[w] = s; red[w + 8] = s2; }
    __syncthreads();
    if (tid == 0) {
        float ts = 0.f, ts2 = 0.f;
        #pragma unroll
        for (int i = 0; i < 8; ++i) { ts += red[i]; ts2 += red[i + 8]; }
        float mu = ts * (1.0f / DD);
        float var = ts2 * (1.0f / DD) - mu * mu;
        stat[0] = mu;
        stat[1] = rsqrtf(var + 1e-5f);
    }
    __syncthreads();
    float mu = stat[0], rs = stat[1];
    float4 gg = ((const float4*)g)[tid];
    float4 bb = ((const float4*)bta)[tid];
    ushort4 ov;
    ov.x = f2bf((v.x - mu) * rs * gg.x + bb.x);
    ov.y = f2bf((v.y - mu) * rs * gg.y + bb.y);
    ov.z = f2bf((v.z - mu) * rs * gg.z + bb.z);
    ov.w = f2bf((v.w - mu) * rs * gg.w + bb.w);
    ((ushort4*)(out + (size_t)t * DD))[tid] = ov;
}

// ---------------- windowed attention: wave per query ----------------
__global__ __launch_bounds__(128)
void attn_kernel(const float* __restrict__ qkv, unsigned short* __restrict__ o)
{
    int wid  = threadIdx.x >> 5;
    int lane = threadIdx.x & 31;
    int flat = blockIdx.x * 4 + wid;           // (b,h,q) flat
    int q = flat & (TT - 1);
    int h = (flat >> 10) & (NH - 1);
    int b = flat >> 14;

    __shared__ float sq[4][HDIM];
    const float* qp = qkv + ((size_t)(b * TT + q)) * (3 * DD) + h * HDIM;
    sq[wid][lane]      = qp[lane];
    sq[wid][lane + 32] = qp[lane + 32];
    __syncthreads();

    int k0 = (q >= WIN - 1) ? q - (WIN - 1) : 0;
    const float scale = 0.125f;                 // 1/sqrt(64)
    float sc[8];
    #pragma unroll
    for (int j = 0; j < 8; ++j) {
        int key = k0 + j * 32 + lane;
        const float* kp = qkv + ((size_t)(b * TT + key)) * (3 * DD) + DD + h * HDIM;
        float acc = 0.f;
        #pragma unroll
        for (int d = 0; d < HDIM; ++d) acc = fmaf(sq[wid][d], kp[d], acc);
        sc[j] = (key <= q) ? acc * scale : -3.0e38f;
    }
    float m = sc[0];
    #pragma unroll
    for (int j = 1; j < 8; ++j) m = fmaxf(m, sc[j]);
    #pragma unroll
    for (int off = 16; off >= 1; off >>= 1) m = fmaxf(m, __shfl_xor(m, off, 32));
    float p[8]; float sum = 0.f;
    #pragma unroll
    for (int j = 0; j < 8; ++j) { p[j] = __expf(sc[j] - m); sum += p[j]; }
    #pragma unroll
    for (int off = 16; off >= 1; off >>= 1) sum += __shfl_xor(sum, off, 32);
    float inv = 1.0f / sum;

    float po[HDIM];
    #pragma unroll
    for (int d = 0; d < HDIM; ++d) po[d] = 0.f;
    #pragma unroll
    for (int j = 0; j < 8; ++j) {
        int key = k0 + j * 32 + lane;
        const float* vp = qkv + ((size_t)(b * TT + key)) * (3 * DD) + 2 * DD + h * HDIM;
        float pj = p[j];
        #pragma unroll
        for (int d = 0; d < HDIM; ++d) po[d] = fmaf(pj, vp[d], po[d]);
    }
    unsigned short* op = o + ((size_t)(b * TT + q)) * DD + h * HDIM;
    #pragma unroll
    for (int d = 0; d < HDIM; ++d) {
        float v = po[d];
        #pragma unroll
        for (int off = 16; off >= 1; off >>= 1) v += __shfl_xor(v, off, 32);
        if (lane == 0) op[d] = f2bf(v * inv);
    }
}

// ---------------- TDM: load a 128 x 32 bf16 tile into LDS (padded rows) ----------
// D# pad: after every 16 DWORDs (one 64B row) insert 4 DWORDs -> LDS row stride
// 80B = 40 ushorts, matching the sA/sB declarations below.
__device__ __forceinline__ void tdm_load_tile(const unsigned short* gptr,
                                              unsigned lds_off, int strideElems)
{
    unsigned long long ga = (unsigned long long)(uintptr_t)gptr;
    su32x4 g0;
    g0[0] = 1u;                                              // count=1, user mode
    g0[1] = lds_off;                                         // lds_addr
    g0[2] = (unsigned)ga;                                    // global_addr[31:0]
    g0[3] = (unsigned)((ga >> 32) & 0x1FFFFFFu) | (2u << 30); // addr[56:32] | type=2
    su32x8 g1;
    g1[0] = (1u << 16)        // data_size = 2 bytes
          | (1u << 20)        // pad_enable
          | (3u << 22)        // pad_interval: 16 DWORDs
          | (3u << 25);       // pad_amount: 4 DWORDs
    g1[1] = ((unsigned)strideElems & 0xFFFFu) << 16;         // tensor_dim0 lo16
    g1[2] = (((unsigned)strideElems >> 16) & 0xFFFFu)        // tensor_dim0 hi16
          | (128u << 16);                                    // tensor_dim1 lo16 = 128
    g1[3] = (32u << 16);                                     // tile_dim0 = 32
    g1[4] = 128u;                                            // tile_dim1 = 128
    g1[5] = (unsigned)strideElems;                           // tensor_dim0_stride lo32
    g1[6] = 0u;
    g1[7] = 0u;
    asm volatile("tensor_load_to_lds %0, %1" :: "s"(g0), "s"(g1) : "memory");
}

// One K-step of WMMAs for a 64x64 wave tile from given LDS buffers.
__device__ __forceinline__ void gemm_step(const unsigned short (&sAb)[128][40],
                                          const unsigned short (&sBb)[128][40],
                                          f32x8 (&acc)[4][4],
                                          int wm, int wn, int fr, int fh)
{
    BF16Frag fa[4], fb[4];
    #pragma unroll
    for (int i = 0; i < 4; ++i) {
        int row = wm * 64 + i * 16 + fr;
        fa[i].u[0] = *(const uint4*)&sAb[row][fh * 8];
        fa[i].u[1] = *(const uint4*)&sAb[row][16 + fh * 8];
    }
    #pragma unroll
    for (int j = 0; j < 4; ++j) {
        int n = wn * 64 + j * 16 + fr;
        fb[j].u[0] = *(const uint4*)&sBb[n][fh * 8];
        fb[j].u[1] = *(const uint4*)&sBb[n][16 + fh * 8];
    }
    #pragma unroll
    for (int i = 0; i < 4; ++i)
        #pragma unroll
        for (int j = 0; j < 4; ++j)
            acc[i][j] = __builtin_amdgcn_wmma_f32_16x16x32_bf16(
                false, fa[i].v, false, fb[j].v, (short)0, acc[i][j], false, false);
}

// ---------------- WMMA GEMM: C[M,N] = A_bf16[M,K] @ W_bf16[N,K]^T ----------------
// Block: 128 threads = 4 waves (2x2), block tile 128x128, wave tile 64x64.
// Double-buffered TDM pipeline: wave0 waits TENSORcnt for the current buffers,
// barrier, then issues the next tile pair into the other buffers (overlapping
// the DMA with this step's WMMAs).
// EPI 0: C = f32 store   EPI 1: C = R + acc (f32)   EPI 2: C = bf16(gelu(acc))
template<int EPI>
__global__ __launch_bounds__(128)
void gemm_kernel(const unsigned short* __restrict__ A, const unsigned short* __restrict__ Wb,
                 float* __restrict__ Cf, unsigned short* __restrict__ Cb,
                 const float* __restrict__ R, int Kk, int ldc)
{
    const int m0 = blockIdx.y * 128;
    const int n0 = blockIdx.x * 128;

    __shared__ __align__(16) unsigned short sA[2][128][40];  // 32 cols + 8 pad (TDM pad)
    __shared__ __align__(16) unsigned short sB[2][128][40];

    const int tid  = threadIdx.x;
    const int lane = tid & 31;
    const int w    = tid >> 5;        // 4 waves
    const int wm   = w >> 1;          // 0..1 -> 64-row slab
    const int wn   = w & 1;           // 0..1 -> 64-col slab
    const int fr   = lane & 15, fh = lane >> 4;

    f32x8 acc[4][4];
    #pragma unroll
    for (int i = 0; i < 4; ++i)
        #pragma unroll
        for (int j = 0; j < 4; ++j)
            #pragma unroll
            for (int e = 0; e < 8; ++e) acc[i][j][e] = 0.0f;

    const unsigned ldsA0 = (unsigned)(uintptr_t)(&sA[0][0][0]);
    const unsigned ldsA1 = (unsigned)(uintptr_t)(&sA[1][0][0]);
    const unsigned ldsB0 = (unsigned)(uintptr_t)(&sB[0][0][0]);
    const unsigned ldsB1 = (unsigned)(uintptr_t)(&sB[1][0][0]);
    const unsigned short* aBase = A  + (size_t)m0 * Kk;
    const unsigned short* bBase = Wb + (size_t)n0 * Kk;

    // prologue: fill buffer 0
    if (w == 0) {
        tdm_load_tile(aBase, ldsA0, Kk);
        tdm_load_tile(bBase, ldsB0, Kk);
    }

    // Kk/32 is even for all GEMMs here (K = 1024 or 2048) -> unroll x2.
    for (int k0 = 0; k0 < Kk; k0 += 64) {
        // ---- phase 0: compute buf0, prefetch into buf1 ----
        if (w == 0) __builtin_amdgcn_s_wait_tensorcnt(0);
        __syncthreads();
        if (w == 0) {                                  // k0+32 < Kk always holds here
            tdm_load_tile(aBase + k0 + 32, ldsA1, Kk);
            tdm_load_tile(bBase + k0 + 32, ldsB1, Kk);
        }
        gemm_step(sA[0], sB[0], acc, wm, wn, fr, fh);

        // ---- phase 1: compute buf1, prefetch into buf0 ----
        if (w == 0) __builtin_amdgcn_s_wait_tensorcnt(0);
        __syncthreads();
        if (w == 0 && k0 + 64 < Kk) {
            tdm_load_tile(aBase + k0 + 64, ldsA0, Kk);
            tdm_load_tile(bBase + k0 + 64, ldsB0, Kk);
        }
        gemm_step(sA[1], sB[1], acc, wm, wn, fr, fh);
    }

    // epilogue: C/D layout -> lane col = lane&15, rows = (lane>>4)*8 + r
    const int colb = n0 + wn * 64 + fr;
    const int rowb = m0 + wm * 64 + fh * 8;
    #pragma unroll
    for (int i = 0; i < 4; ++i) {
        #pragma unroll
        for (int j = 0; j < 4; ++j) {
            int col = colb + j * 16;
            #pragma unroll
            for (int r = 0; r < 8; ++r) {
                int row = rowb + i * 16 + r;
                size_t off = (size_t)row * ldc + col;
                float v = acc[i][j][r];
                if (EPI == 0) {
                    Cf[off] = v;
                } else if (EPI == 1) {
                    Cf[off] = R[off] + v;
                } else {
                    float g = 0.5f * v * (1.0f + erff(v * 0.70710678118654752f));
                    Cb[off] = f2bf(g);
                }
            }
        }
    }
}

// ---------------- driver ----------------
extern "C" void kernel_launch(void* const* d_in, const int* in_sizes, int n_in,
                              void* d_out, int out_size, void* d_ws, size_t ws_size,
                              hipStream_t stream)
{
    const int*   idx   = (const int*)d_in[0];
    const float* tok   = (const float*)d_in[1];
    const float* pos   = (const float*)d_in[2];
    const float* ln1g  = (const float*)d_in[3];
    const float* ln1b  = (const float*)d_in[4];
    const float* qkvw  = (const float*)d_in[5];
    const float* outw  = (const float*)d_in[6];
    const float* ln2g  = (const float*)d_in[7];
    const float* ln2b  = (const float*)d_in[8];
    const float* aw1   = (const float*)d_in[9];
    const float* aw2   = (const float*)d_in[10];
    const float* fw1   = (const float*)d_in[11];
    const float* fw2   = (const float*)d_in[12];
    const float* lnfg  = (const float*)d_in[13];
    const float* lnfb  = (const float*)d_in[14];
    const float* headw = (const float*)d_in[15];

    char* ws = (char*)d_ws;
    float*          x   = (float*)(ws);                               // 8 MB
    unsigned short* hb  = (unsigned short*)(ws + ((size_t)8  << 20)); // 4 MB (bf16 LN out)
    float*          qkv = (float*)(ws + ((size_t)12 << 20));          // 24 MB
    unsigned short* ob  = (unsigned short*)(ws + ((size_t)36 << 20)); // 4 MB (bf16 attn out)
    unsigned short* g1  = (unsigned short*)(ws + ((size_t)40 << 20)); // 8 MB
    unsigned short* g2  = (unsigned short*)(ws + ((size_t)48 << 20)); // 8 MB
    unsigned short* wb  = (unsigned short*)(ws + ((size_t)56 << 20)); // 64 MB bf16 weights

    const int TD = 3 * DD;   // 3072

    embed_kernel<<<MTOK, 256, 0, stream>>>(idx, tok, pos, x);

    for (int l = 0; l < NL; ++l) {
        // ---- attention ----
        ln_kernel<<<MTOK, 256, 0, stream>>>(x, ln1g + (size_t)l * DD, ln1b + (size_t)l * DD, hb);
        cvt_kernel<<<(TD * DD) / 2048, 256, 0, stream>>>(qkvw + (size_t)l * TD * DD, wb);
        gemm_kernel<0><<<dim3(TD / 128, MTOK / 128), 128, 0, stream>>>(
            hb, wb, qkv, nullptr, nullptr, DD, TD);
        attn_kernel<<<(BB * NH * TT) / 4, 128, 0, stream>>>(qkv, ob);
        cvt_kernel<<<(DD * DD) / 2048, 256, 0, stream>>>(outw + (size_t)l * DD * DD, wb);
        gemm_kernel<1><<<dim3(DD / 128, MTOK / 128), 128, 0, stream>>>(
            ob, wb, x, nullptr, x, DD, DD);
        // ---- MLP (two gelu branches) ----
        ln_kernel<<<MTOK, 256, 0, stream>>>(x, ln2g + (size_t)l * DD, ln2b + (size_t)l * DD, hb);
        cvt_kernel<<<(HALFF * DD) / 2048, 256, 0, stream>>>(aw1 + (size_t)l * HALFF * DD, wb);
        gemm_kernel<2><<<dim3(HALFF / 128, MTOK / 128), 128, 0, stream>>>(
            hb, wb, nullptr, g1, nullptr, DD, HALFF);
        cvt_kernel<<<(HALFF * DD) / 2048, 256, 0, stream>>>(fw1 + (size_t)l * HALFF * DD, wb);
        gemm_kernel<2><<<dim3(HALFF / 128, MTOK / 128), 128, 0, stream>>>(
            hb, wb, nullptr, g2, nullptr, DD, HALFF);
        cvt_kernel<<<(DD * HALFF) / 2048, 256, 0, stream>>>(aw2 + (size_t)l * DD * HALFF, wb);
        gemm_kernel<1><<<dim3(DD / 128, MTOK / 128), 128, 0, stream>>>(
            g1, wb, x, nullptr, x, HALFF, DD);
        cvt_kernel<<<(DD * HALFF) / 2048, 256, 0, stream>>>(fw2 + (size_t)l * DD * HALFF, wb);
        gemm_kernel<1><<<dim3(DD / 128, MTOK / 128), 128, 0, stream>>>(
            g2, wb, x, nullptr, x, HALFF, DD);
    }

    ln_kernel<<<MTOK, 256, 0, stream>>>(x, lnfg, lnfb, hb);
    cvt_kernel<<<((size_t)VV * DD) / 2048, 256, 0, stream>>>(headw, wb);
    gemm_kernel<0><<<dim3(VV / 128, MTOK / 128), 128, 0, stream>>>(
        hb, wb, (float*)d_out, nullptr, nullptr, DD, VV);
}
